// MinGRUModel_35459249996095
// MI455X (gfx1250) — matrix operand order repaired and checked by reading, not compiled, not run on hardware
//
#include <hip/hip_runtime.h>
#include <hip/hip_bf16.h>

// ---------- types for WMMA ----------
typedef __attribute__((ext_vector_type(8)))  float  v8f;
typedef __attribute__((ext_vector_type(4)))  __bf16 bf16x4;
typedef __attribute__((ext_vector_type(8)))  __bf16 bf16x8;
typedef __attribute__((ext_vector_type(16))) __bf16 bf16x16;

union FragU { bf16x16 f; bf16x8 h[2]; };

__device__ inline bf16x16 make_frag(const __bf16* p0, const __bf16* p1) {
    FragU u;
    u.h[0] = *(const bf16x8*)p0;   // 16B -> ds_load_b128
    u.h[1] = *(const bf16x8*)p1;   // 16B -> ds_load_b128
    return u.f;
}

__device__ inline float sigmoidf_dev(float x) {
    return 1.0f / (1.0f + __expf(-x));
}

// ---------- problem constants ----------
#define BDIM   8
#define SLEN   4096
#define DDIM   1024
#define MROWS  (BDIM * SLEN)     // 32768

#define BM 64
#define BN 64
#define BK 32

#define CHUNKS 16
#define CS     (SLEN / CHUNKS)   // 256

// LDS layout (bytes), single __shared__ object => workgroup LDS offset 0:
//   buf b (b=0,1) at b*16384 ; within buf: A tile @ +0, B tile m3 @ +(m3+1)*4096
#define LDS_BUF_BYTES   16384u
#define LDS_TILE_BYTES  4096u
#define SMEM_ELEMS      (2 * 4 * BM * BK)   // 16384 __bf16 = 32 KB

// ============================================================
// Kernel 0: fp32 -> bf16 conversion (x and the three W matrices)
// ============================================================
__global__ __launch_bounds__(256)
void cvt_f32_to_bf16(const float* __restrict__ src, __bf16* __restrict__ dst, int n4)
{
    const int i = blockIdx.x * blockDim.x + threadIdx.x;
    if (i >= n4) return;
    const float4 f = ((const float4*)src)[i];
    bf16x4 o;
    o.x = (__bf16)f.x; o.y = (__bf16)f.y;
    o.z = (__bf16)f.z; o.w = (__bf16)f.w;
    ((bf16x4*)dst)[i] = o;
}

// ============================================================
// Async staging of one K-step: A tile (64x32 bf16) + 3 B tiles.
// 8 x global_load_async_to_lds_b128 per thread per stage.
// The generic pointer to the LDS buffer is passed as a (unreferenced)
// asm operand so LLVM treats the LDS object as escaped/written by the
// asm (otherwise the ds_load fragment reads get folded to undef).
// ============================================================
__device__ inline void stage_tiles_async(const __bf16* __restrict__ xg,
                                         const __bf16* __restrict__ w0,
                                         const __bf16* __restrict__ w1,
                                         const __bf16* __restrict__ w2,
                                         int mm, int nn, int kk, int K,
                                         unsigned ldsBufBase, int tid,
                                         const __bf16* smemPtr)
{
    const unsigned r    = (unsigned)(tid >> 2);   // 0..31
    const unsigned sg   = (unsigned)(tid & 3);    // 16B segment of a 64B row
    const unsigned rowB = (unsigned)K * 2u;       // global row stride in bytes

    // --- A tile ---
    {
        const unsigned long long base =
            (unsigned long long)(xg + (size_t)mm * K + kk);
        #pragma unroll
        for (int j = 0; j < 2; ++j) {
            const unsigned rr = r + (unsigned)j * 32u;
            const unsigned go = rr * rowB + sg * 16u;
            const unsigned lo = ldsBufBase + rr * 64u + sg * 16u;
            asm volatile("global_load_async_to_lds_b128 %0, %1, %2"
                         :: "v"(lo), "v"(go), "s"(base), "s"(smemPtr)
                         : "memory");
        }
    }
    // --- B tiles (Wg, Wd, Wv) ---
    const __bf16* Ws[3] = { w0, w1, w2 };
    #pragma unroll
    for (int m3 = 0; m3 < 3; ++m3) {
        const unsigned long long base =
            (unsigned long long)(Ws[m3] + (size_t)nn * K + kk);
        const unsigned lbase = ldsBufBase + (unsigned)(m3 + 1) * LDS_TILE_BYTES;
        #pragma unroll
        for (int j = 0; j < 2; ++j) {
            const unsigned rr = r + (unsigned)j * 32u;
            const unsigned go = rr * rowB + sg * 16u;
            const unsigned lo = lbase + rr * 64u + sg * 16u;
            asm volatile("global_load_async_to_lds_b128 %0, %1, %2"
                         :: "v"(lo), "v"(go), "s"(base), "s"(smemPtr)
                         : "memory");
        }
    }
}

// ============================================================
// Kernel 1: fused 3-matrix projection GEMM (bf16 WMMA, fp32 acc),
// double-buffered LDS fed by async global->LDS DMA.
//   g = sigmoid(x*Wg^T + bg)
//   a = sigmoid(x*Wd^T + bd)
//   u = (1-a) * (x*Wv^T + bv)
// ============================================================
__global__ __launch_bounds__(128)
void mingru_gemm3(const __bf16* __restrict__ x16,
                  const __bf16* __restrict__ Wg,
                  const __bf16* __restrict__ Wd,
                  const __bf16* __restrict__ Wv,
                  const float* __restrict__ bg,
                  const float* __restrict__ bd,
                  const float* __restrict__ bv,
                  float* __restrict__ gBuf,
                  float* __restrict__ aBuf,
                  float* __restrict__ uBuf,
                  int M, int K, int N)
{
    __shared__ __align__(16) __bf16 smem[SMEM_ELEMS];   // 32 KB, LDS offset 0

    // Escape the LDS buffer so asm with memory clobbers is assumed to write it.
    const __bf16* smemPtr = smem;
    asm volatile("" :: "s"(smemPtr) : "memory");

    const int nBlocks = N / BN;
    const int by = blockIdx.x / nBlocks;
    const int bx = blockIdx.x % nBlocks;
    const int mm = by * BM;
    const int nn = bx * BN;

    const int tid  = threadIdx.x;
    const int lane = tid & 31;
    const int w    = tid >> 5;            // 4 waves
    const int wm   = (w >> 1) * 32;       // wave row offset in tile
    const int wn   = (w & 1)  * 32;       // wave col offset in tile

    v8f acc[3][2][2] = {};                // [matrix][tileM][tileN]

    const int rl   = lane & 15;
    const int kSel = (lane >> 4) * 8;     // A frag K select
    const int kSb  = (lane >> 4) * 16;    // unused; B frags also use kSel

    // prologue: stage k=0 into buf 0
    stage_tiles_async(x16, Wg, Wd, Wv, mm, nn, 0, K, 0u, tid, smemPtr);

    const int nSteps = K / BK;
    for (int it = 0; it < nSteps; ++it) {
        const int cur = it & 1;
        const int kkn = (it + 1) * BK;
        if (kkn < K) {
            stage_tiles_async(x16, Wg, Wd, Wv, mm, nn, kkn, K,
                              (unsigned)(cur ^ 1) * LDS_BUF_BYTES, tid, smemPtr);
            // current stage's 8 loads are the oldest outstanding -> wait <= 8
            asm volatile("s_wait_asynccnt 0x8" :: "s"(smemPtr) : "memory");
        } else {
            asm volatile("s_wait_asynccnt 0x0" :: "s"(smemPtr) : "memory");
        }
        __syncthreads();   // all waves' async data for this stage visible

        const __bf16* sbuf = smem + (size_t)cur * (LDS_BUF_BYTES / 2); // elems

        // A fragments for the two 16-row tiles of this wave
        bf16x16 aF[2];
        #pragma unroll
        for (int tm = 0; tm < 2; ++tm) {
            const __bf16* row = sbuf + (wm + tm * 16 + rl) * BK;
            aF[tm] = make_frag(row + kSel, row + 16 + kSel);
        }

        #pragma unroll
        for (int m3 = 0; m3 < 3; ++m3) {
            const __bf16* sB = sbuf + (m3 + 1) * (BN * BK);
            #pragma unroll
            for (int tn = 0; tn < 2; ++tn) {
                const __bf16* row = sB + (wn + tn * 16 + rl) * BK;
                bf16x16 bF = make_frag(row + kSel, row + 16 + kSel);
                #pragma unroll
                for (int tm = 0; tm < 2; ++tm) {
                    acc[m3][tm][tn] = __builtin_amdgcn_wmma_f32_16x16x32_bf16(
                        false, aF[tm], false, bF,
                        (short)0, acc[m3][tm][tn], false, false);
                }
            }
        }
        __syncthreads();   // done reading buf[cur]; safe to refill it next iter
    }

    // --- epilogue: activations + write g/a/u ---
    const int rHi = (lane >> 4) * 8;
    #pragma unroll
    for (int tn = 0; tn < 2; ++tn) {
        const int n = nn + wn + tn * 16 + rl;
        const float vbg = bg[n];
        const float vbd = bd[n];
        const float vbv = bv[n];
        #pragma unroll
        for (int tm = 0; tm < 2; ++tm) {
            #pragma unroll
            for (int i = 0; i < 8; ++i) {
                const int m = mm + wm + tm * 16 + rHi + i;
                const size_t o = (size_t)m * N + n;
                const float gv = sigmoidf_dev(acc[0][tm][tn][i] + vbg);
                const float av = sigmoidf_dev(acc[1][tm][tn][i] + vbd);
                const float uv = (1.0f - av) * (acc[2][tm][tn][i] + vbv);
                gBuf[o] = gv;
                aBuf[o] = av;
                uBuf[o] = uv;
            }
        }
    }
}

// ============================================================
// Kernel 2: per-chunk local scan.  One thread per (b, chunk, e).
// ============================================================
__global__ __launch_bounds__(256)
void scan_pass1(const float* __restrict__ aBuf,
                const float* __restrict__ uBuf,
                float* __restrict__ carryA,
                float* __restrict__ carryH)
{
    const int gid = blockIdx.x * blockDim.x + threadIdx.x;   // ((b*CHUNKS + c)*D + e)
    const int e  = gid % DDIM;
    const int bc = gid / DDIM;
    const int c  = bc % CHUNKS;
    const int b  = bc / CHUNKS;
    if (b >= BDIM) return;

    const size_t base = ((size_t)b * SLEN + (size_t)c * CS) * DDIM + e;
    float A = 1.0f, h = 0.0f;
    for (int i = 0; i < CS; ++i) {
        const size_t idx = base + (size_t)i * DDIM;
        if (i + 8 < CS) {
            __builtin_prefetch(aBuf + idx + 8 * DDIM, 0, 0);
            __builtin_prefetch(uBuf + idx + 8 * DDIM, 0, 0);
        }
        const float av = aBuf[idx];
        const float uv = uBuf[idx];
        h = av * h + uv;
        A *= av;
    }
    carryA[gid] = A;
    carryH[gid] = h;
}

// ============================================================
// Kernel 3: combine chunk carries per channel (16-step affine scan).
// ============================================================
__global__ __launch_bounds__(256)
void scan_pass2(const float* __restrict__ carryA,
                const float* __restrict__ carryH,
                float* __restrict__ carryIn)
{
    const int gid = blockIdx.x * blockDim.x + threadIdx.x;   // b*D + e
    const int e = gid % DDIM;
    const int b = gid / DDIM;
    if (b >= BDIM) return;

    float h = 0.0f;
    #pragma unroll
    for (int c = 0; c < CHUNKS; ++c) {
        const size_t o = ((size_t)b * CHUNKS + c) * DDIM + e;
        carryIn[o] = h;
        h = carryA[o] * h + carryH[o];
    }
}

// ============================================================
// Kernel 4: rescan with carry-in, fuse out = h * g.
// ============================================================
__global__ __launch_bounds__(256)
void scan_pass3(const float* __restrict__ aBuf,
                const float* __restrict__ uBuf,
                const float* __restrict__ gBuf,
                const float* __restrict__ carryIn,
                float* __restrict__ out)
{
    const int gid = blockIdx.x * blockDim.x + threadIdx.x;   // ((b*CHUNKS + c)*D + e)
    const int e  = gid % DDIM;
    const int bc = gid / DDIM;
    const int c  = bc % CHUNKS;
    const int b  = bc / CHUNKS;
    if (b >= BDIM) return;

    const size_t base = ((size_t)b * SLEN + (size_t)c * CS) * DDIM + e;
    float h = carryIn[gid];
    for (int i = 0; i < CS; ++i) {
        const size_t idx = base + (size_t)i * DDIM;
        if (i + 8 < CS) {
            __builtin_prefetch(aBuf + idx + 8 * DDIM, 0, 0);
            __builtin_prefetch(uBuf + idx + 8 * DDIM, 0, 0);
            __builtin_prefetch(gBuf + idx + 8 * DDIM, 0, 0);
        }
        const float av = aBuf[idx];
        const float uv = uBuf[idx];
        h = av * h + uv;
        out[idx] = h * gBuf[idx];
    }
}

// ============================================================
// host-side launch
// ============================================================
extern "C" void kernel_launch(void* const* d_in, const int* in_sizes, int n_in,
                              void* d_out, int out_size, void* d_ws, size_t ws_size,
                              hipStream_t stream)
{
    (void)in_sizes; (void)n_in; (void)out_size; (void)ws_size;

    const float* x  = (const float*)d_in[0];
    const float* Wg = (const float*)d_in[1];
    const float* bg = (const float*)d_in[2];
    const float* Wv = (const float*)d_in[3];
    const float* bv = (const float*)d_in[4];
    const float* Wd = (const float*)d_in[5];
    const float* bd = (const float*)d_in[6];
    float* out = (float*)d_out;

    const int M = MROWS;        // 32768
    const int K = DDIM;         // 1024
    const int N = DDIM;         // 1024

    char* ws = (char*)d_ws;
    const size_t per = (size_t)M * N * sizeof(float);     // 128 MB
    float* uBuf    = (float*)(ws);
    float* aBuf    = (float*)(ws + per);
    float* gBuf    = (float*)(ws + 2 * per);
    float* carryA  = (float*)(ws + 3 * per);
    float* carryH  = carryA + (size_t)BDIM * CHUNKS * DDIM;
    float* carryIn = carryH + (size_t)BDIM * CHUNKS * DDIM;
    char*  bfBase  = (char*)(carryIn + (size_t)BDIM * CHUNKS * DDIM);
    __bf16* x16  = (__bf16*)bfBase;                               // 64 MB
    __bf16* Wg16 = x16  + (size_t)M * K;
    __bf16* Wd16 = Wg16 + (size_t)K * N;
    __bf16* Wv16 = Wd16 + (size_t)K * N;

    // 0) one-time fp32 -> bf16 conversion (x + three weight matrices)
    {
        const int n4x = (M * K) / 4;          // 8,388,608
        cvt_f32_to_bf16<<<(n4x + 255) / 256, 256, 0, stream>>>(x, x16, n4x);
        const int n4w = (K * N) / 4;          // 262,144
        cvt_f32_to_bf16<<<(n4w + 255) / 256, 256, 0, stream>>>(Wg, Wg16, n4w);
        cvt_f32_to_bf16<<<(n4w + 255) / 256, 256, 0, stream>>>(Wd, Wd16, n4w);
        cvt_f32_to_bf16<<<(n4w + 255) / 256, 256, 0, stream>>>(Wv, Wv16, n4w);
    }

    // 1) fused triple projection GEMM (WMMA bf16, async LDS staging)
    {
        dim3 grid((M / BM) * (N / BN));   // 512*16 = 8192 blocks
        dim3 block(128);
        mingru_gemm3<<<grid, block, 0, stream>>>(x16, Wg16, Wd16, Wv16,
                                                 bg, bd, bv,
                                                 gBuf, aBuf, uBuf, M, K, N);
    }
    // 2) chunked causal scan
    {
        const int nThreads = BDIM * CHUNKS * DDIM;        // 131072
        scan_pass1<<<nThreads / 256, 256, 0, stream>>>(aBuf, uBuf, carryA, carryH);
    }
    {
        const int nThreads = BDIM * DDIM;                 // 8192
        scan_pass2<<<nThreads / 256, 256, 0, stream>>>(carryA, carryH, carryIn);
    }
    {
        const int nThreads = BDIM * CHUNKS * DDIM;        // 131072
        scan_pass3<<<nThreads / 256, 256, 0, stream>>>(aBuf, uBuf, gBuf, carryIn, out);
    }
}
